// ann2_poisson_snn1_29171417874459
// MI455X (gfx1250) — matrix operand order repaired and checked by reading, not compiled, not run on hardware
//
#include <hip/hip_runtime.h>
#include <cmath>

typedef _Float16 v16h __attribute__((ext_vector_type(16)));
typedef _Float16 v8h  __attribute__((ext_vector_type(8)));
typedef float    v8f  __attribute__((ext_vector_type(8)));
typedef int      v4i  __attribute__((ext_vector_type(4)));

#define AS1 __attribute__((address_space(1)))
#define AS3 __attribute__((address_space(3)))

#if __has_builtin(__builtin_amdgcn_global_load_async_to_lds_b128)
#define HAVE_ASYNC_LDS 1
#endif

// Load 16 contiguous halfs (32B) as two aligned 16B chunks -> v16h fragment.
__device__ __forceinline__ v16h load_frag16(const _Float16* p) {
    v8h lo = *(const v8h*)(p);
    v8h hi = *(const v8h*)(p + 8);
    return __builtin_shufflevector(lo, hi, 0,1,2,3,4,5,6,7,8,9,10,11,12,13,14,15);
}

// ---------------------------------------------------------------------------
// K0a: pre-swizzle w1 (500x784 f32) into f16 A-fragments.
// 32 M-tiles (512 rows), 25 K-steps (800, zero-padded).
// A layout: row M = lane&15, hi = lane>>4, k(h) = hi*16+h (matches B fragment)
// ---------------------------------------------------------------------------
__global__ __launch_bounds__(256) void swz_w1_kernel(const float* __restrict__ w1,
                                                     _Float16* __restrict__ out) {
    int tid = blockIdx.x * 256 + threadIdx.x;
    if (tid >= 32 * 25 * 32) return;
    int lane = tid & 31;
    int rest = tid >> 5;
    int kt = rest % 25;
    int mt = rest / 25;
    int row = mt * 16 + (lane & 15);
    int hi  = lane >> 4;
    _Float16* o = out + (size_t)tid * 16;
#pragma unroll
    for (int h = 0; h < 16; ++h) {
        int k = kt * 32 + hi * 16 + h;
        float v = (row < 500 && k < 784) ? w1[row * 784 + k] : 0.0f;
        o[h] = (_Float16)v;
    }
}

// ---------------------------------------------------------------------------
// K0b: pre-swizzle w2 (1000x500 f32). Swizzled rows 0..511 = mu rows 0..499,
// rows 512..1023 = ln_var rows 500..999 (padding rows zero). 64 M-tiles,
// 16 K-steps (512, zero-padded past 500).
// ---------------------------------------------------------------------------
__global__ __launch_bounds__(256) void swz_w2_kernel(const float* __restrict__ w2,
                                                     _Float16* __restrict__ out) {
    int tid = blockIdx.x * 256 + threadIdx.x;
    if (tid >= 64 * 16 * 32) return;
    int lane = tid & 31;
    int rest = tid >> 5;
    int kt = rest % 16;
    int mt = rest / 16;
    int ms = mt * 16 + (lane & 15);
    int hi = lane >> 4;
    int src;
    if (ms < 512) src = (ms < 500) ? ms : -1;
    else { int f = ms - 512; src = (f < 500) ? (500 + f) : -1; }
    _Float16* o = out + (size_t)tid * 16;
#pragma unroll
    for (int h = 0; h < 16; ++h) {
        int k = kt * 32 + hi * 16 + h;
        float v = (src >= 0 && k < 500) ? w2[src * 500 + k] : 0.0f;
        o[h] = (_Float16)v;
    }
}

// ---------------------------------------------------------------------------
// K1: GEMM1  h1[b] = relu(w1 @ inputs[b] + b1), f16 out (b,512,100).
// Block = 8 waves; each wave owns FOUR 16x16 M-tiles -> block covers all
// 512 M rows, so each staged B tile (and each input element) is read once.
// grid = (7 t-tiles, 256 batches). K = 25 steps of 32.
// B tile staged f32->f16 into LDS [n][k], stride 40 halfs, double-buffered.
// ---------------------------------------------------------------------------
__global__ __launch_bounds__(256) void gemm1_kernel(const float* __restrict__ x,
                                                    const float* __restrict__ b1,
                                                    const _Float16* __restrict__ w1s,
                                                    _Float16* __restrict__ h1) {
    const int KT = 25;
    __shared__ __align__(32) _Float16 bt[2][16 * 40];
    const int tt = blockIdx.x, b = blockIdx.y;
    const int t0 = tt * 16;
    const int lane = threadIdx.x & 31, wave = threadIdx.x >> 5;
    const int n = lane & 15, hl = lane >> 4;
    const float* xb = x + (size_t)b * (784 * 100);

    auto stage = [&](int kt, int buf) {
#pragma unroll
        for (int e = 0; e < 2; ++e) {
            int idx = threadIdx.x + e * 256;
            int k = idx >> 4, c = idx & 15;
            int kk = kt * 32 + k, t = t0 + c;
            float v = (kk < 784 && t < 100) ? xb[kk * 100 + t] : 0.0f;
            bt[buf][c * 40 + k] = (_Float16)v;
        }
    };

    stage(0, 0);

    // A fragment base: tiles mt = wave*4 + j, consecutive in j.
    const _Float16* aBase = w1s + ((size_t)(wave * 4) * KT * 32 + lane) * 16;
    const v8f vzero = {};
    v8f acc[4];
#pragma unroll
    for (int j = 0; j < 4; ++j) acc[j] = vzero;

    for (int kt = 0; kt < KT; ++kt) {
        __syncthreads();
        if (kt + 1 < KT) {
            stage(kt + 1, (kt + 1) & 1);
#pragma unroll
            for (int j = 0; j < 4; ++j)
                __builtin_prefetch(aBase + ((size_t)j * KT + kt + 1) * 512, 0, 1);
        }
        v16h bf = load_frag16(&bt[kt & 1][n * 40 + hl * 16]);
#pragma unroll
        for (int j = 0; j < 4; ++j) {
            v16h a = load_frag16(aBase + ((size_t)j * KT + kt) * 512);
            acc[j] = __builtin_amdgcn_wmma_f32_16x16x32_f16(false, a, false, bf,
                                                            (short)0, acc[j], false, false);
        }
    }

    int t = t0 + n;
    if (t < 100) {
        int hi8 = hl * 8;
        _Float16* outp = h1 + (size_t)b * (512 * 100);
#pragma unroll
        for (int j = 0; j < 4; ++j) {
            int m0 = (wave * 4 + j) * 16;
#pragma unroll
            for (int r = 0; r < 8; ++r) {
                int m = m0 + r + hi8;
                float bias = (m < 500) ? b1[m] : 0.0f;
                float v = acc[j][r] + bias;
                v = v > 0.0f ? v : 0.0f;
                outp[m * 100 + t] = (_Float16)v;
            }
        }
    }
}

// ---------------------------------------------------------------------------
// K2: GEMM2 + fused reparameterization.
// Waves 0-3 compute mu tiles (8 each -> rows 0..511), waves 4-7 the matching
// ln_var tiles (swizzled rows 512..1023); each staged h1 tile is read once.
// ln_var accumulators exchanged via LDS; epilogue computes
// sample = relu(mu+b2[f]) + eps * exp(0.5*relu(lv+b2[500+f])).
// grid = (7 t-tiles, 256 batches). K = 16 steps of 32.
// ---------------------------------------------------------------------------
__global__ __launch_bounds__(256) void gemm2_kernel(const _Float16* __restrict__ h1,
                                                    const float* __restrict__ b2,
                                                    const _Float16* __restrict__ w2s,
                                                    const float* __restrict__ eps,
                                                    float* __restrict__ sample) {
    const int KT = 16;
    __shared__ __align__(32) _Float16 bt[2][16 * 40];
    __shared__ float xfr[32][16 * 18];
    const int tt = blockIdx.x, b = blockIdx.y;
    const int t0 = tt * 16;
    const int lane = threadIdx.x & 31, wave = threadIdx.x >> 5;
    const int w4 = wave & 3, side = wave >> 2;
    const int n = lane & 15, hl = lane >> 4;
    const _Float16* hb = h1 + (size_t)b * (512 * 100);

    auto stage = [&](int kt, int buf) {
#pragma unroll
        for (int e = 0; e < 2; ++e) {
            int idx = threadIdx.x + e * 256;
            int k = idx >> 4, c = idx & 15;
            int t = t0 + c;
            bt[buf][c * 40 + k] = (t < 100) ? hb[(kt * 32 + k) * 100 + t] : (_Float16)0.0f;
        }
    };

    stage(0, 0);

    // Tiles: tid8 = w4*8 + j (0..31); swizzled mt = side*32 + tid8, consecutive in j.
    const _Float16* aBase = w2s + ((size_t)(side * 32 + w4 * 8) * KT * 32 + lane) * 16;
    const v8f vzero = {};
    v8f acc[8];
#pragma unroll
    for (int j = 0; j < 8; ++j) acc[j] = vzero;

    for (int kt = 0; kt < KT; ++kt) {
        __syncthreads();
        if (kt + 1 < KT) {
            stage(kt + 1, (kt + 1) & 1);
#pragma unroll
            for (int j = 0; j < 8; ++j)
                __builtin_prefetch(aBase + ((size_t)j * KT + kt + 1) * 512, 0, 1);
        }
        v16h bf = load_frag16(&bt[kt & 1][n * 40 + hl * 16]);
#pragma unroll
        for (int j = 0; j < 8; ++j) {
            v16h a = load_frag16(aBase + ((size_t)j * KT + kt) * 512);
            acc[j] = __builtin_amdgcn_wmma_f32_16x16x32_f16(false, a, false, bf,
                                                            (short)0, acc[j], false, false);
        }
    }

    int hi8 = hl * 8;
    if (side == 1) {
#pragma unroll
        for (int j = 0; j < 8; ++j) {
            int tid8 = w4 * 8 + j;
#pragma unroll
            for (int r = 0; r < 8; ++r) xfr[tid8][(r + hi8) * 18 + n] = acc[j][r];
        }
    }
    __syncthreads();
    if (side == 0) {
        int t = t0 + n;
        if (t < 100) {
#pragma unroll
            for (int j = 0; j < 8; ++j) {
                int tid8 = w4 * 8 + j;
                int f0 = tid8 * 16;
#pragma unroll
                for (int r = 0; r < 8; ++r) {
                    int m = r + hi8;
                    int f = f0 + m;
                    if (f < 500) {
                        float mu = acc[j][r] + b2[f];
                        mu = mu > 0.0f ? mu : 0.0f;
                        float lv = xfr[tid8][m * 18 + n] + b2[500 + f];
                        lv = lv > 0.0f ? lv : 0.0f;
                        size_t o = (size_t)b * 50000 + (size_t)f * 100 + t;
                        sample[o] = mu + eps[o] * __expf(0.5f * lv);
                    }
                }
            }
        }
    }
}

// ---------------------------------------------------------------------------
// K3: axon scan. One thread per (b,f) chain: u_t = a1*u_{t-1} + a2*u_{t-2}
// + V0*x_t. Reads sample (b,500,100) via float4; writes psp (b,100,512).
// ---------------------------------------------------------------------------
__global__ __launch_bounds__(256) void axon_kernel(const float* __restrict__ sample,
                                                   const float* __restrict__ alpha1,
                                                   const float* __restrict__ alpha2,
                                                   float v0c,
                                                   float* __restrict__ psp) {
    int idx = blockIdx.x * 256 + threadIdx.x;
    if (idx >= 256 * 500) return;
    int b = idx / 500, f = idx % 500;
    float a1 = alpha1[f], a2 = alpha2[f];
    const float4* src = (const float4*)(sample + (size_t)b * 50000 + (size_t)f * 100);
    float* dst = psp + (size_t)b * 100 * 512 + f;
    float u1 = 0.0f, u2 = 0.0f;
    for (int tq = 0; tq < 25; ++tq) {
        float4 x4 = src[tq];
        float xs[4] = {x4.x, x4.y, x4.z, x4.w};
#pragma unroll
        for (int j = 0; j < 4; ++j) {
            float u = a1 * u1 + a2 * u2 + v0c * xs[j];
            dst[(size_t)(tq * 4 + j) * 512] = u;
            u2 = u1; u1 = u;
        }
    }
}

// ---------------------------------------------------------------------------
// K4: output projection. One wave per (b,t): 10 dots of length 500 with w3
// staged in LDS (async DMA when available), butterfly-reduced across wave32.
// cur layout: (b,100,16) (o padded to 16).
// ---------------------------------------------------------------------------
__global__ __launch_bounds__(256) void cur_kernel(const float* __restrict__ psp,
                                                  const float* __restrict__ w3,
                                                  const float* __restrict__ b3,
                                                  float* __restrict__ cur) {
    __shared__ __align__(16) float w3s[5000];
#ifdef HAVE_ASYNC_LDS
    // 10*500 floats = 20000 B = 1250 x 16B chunks, DMA'd to LDS.
    for (int i = threadIdx.x; i < 1250; i += 256) {
        __builtin_amdgcn_global_load_async_to_lds_b128(
            (AS1 v4i*)(w3 + i * 4), (AS3 v4i*)(&w3s[i * 4]), 0, 0);
    }
#if __has_builtin(__builtin_amdgcn_s_wait_asynccnt)
    __builtin_amdgcn_s_wait_asynccnt(0);
#else
    asm volatile("s_wait_asynccnt 0" ::: "memory");
#endif
#else
    for (int i = threadIdx.x; i < 5000; i += 256) w3s[i] = w3[i];
#endif
    __syncthreads();

    int gw = blockIdx.x * 8 + (threadIdx.x >> 5);
    int lane = threadIdx.x & 31;
    if (gw >= 25600) return;
    int b = gw / 100, t = gw % 100;
    const float* p = psp + ((size_t)b * 100 + t) * 512;
    float acc[10];
#pragma unroll
    for (int o = 0; o < 10; ++o) acc[o] = 0.0f;
    for (int f = lane; f < 500; f += 32) {
        float pv = p[f];
#pragma unroll
        for (int o = 0; o < 10; ++o) acc[o] += pv * w3s[o * 500 + f];
    }
#pragma unroll
    for (int o = 0; o < 10; ++o) {
        float v = acc[o];
#pragma unroll
        for (int m = 16; m >= 1; m >>= 1) v += __shfl_xor(v, m, 32);
        if (lane == 0) cur[((size_t)b * 100 + t) * 16 + o] = v + b3[o];
    }
}

// ---------------------------------------------------------------------------
// K5: LIF scan + spike threshold. One thread per (b,o). Output (B,10,T) f32.
// ---------------------------------------------------------------------------
__global__ __launch_bounds__(256) void lif_kernel(const float* __restrict__ cur,
                                                  float am,
                                                  float* __restrict__ out) {
    int idx = blockIdx.x * 256 + threadIdx.x;
    if (idx >= 2560) return;
    int b = idx / 10, o = idx % 10;
    const float* c = cur + (size_t)b * 100 * 16 + o;
    float* op = out + ((size_t)b * 10 + o) * 100;
    float v = 0.0f, s = 0.0f;
    for (int t = 0; t < 100; ++t) {
        v = am * v * (1.0f - s) + c[(size_t)t * 16];
        s = (v > 1.0f) ? 1.0f : 0.0f;
        op[t] = s;
    }
}

// ---------------------------------------------------------------------------
extern "C" void kernel_launch(void* const* d_in, const int* in_sizes, int n_in,
                              void* d_out, int out_size, void* d_ws, size_t ws_size,
                              hipStream_t stream) {
    (void)in_sizes; (void)n_in; (void)out_size;
    const float* x      = (const float*)d_in[0];   // (256,784,100)
    const float* eps    = (const float*)d_in[1];   // (256,500,100)
    const float* w1     = (const float*)d_in[2];   // (500,784)
    const float* b1     = (const float*)d_in[3];   // (500,)
    const float* w2     = (const float*)d_in[4];   // (1000,500)
    const float* b2     = (const float*)d_in[5];   // (1000,)
    const float* alpha1 = (const float*)d_in[6];   // (500,)
    const float* alpha2 = (const float*)d_in[7];   // (500,)
    const float* w3     = (const float*)d_in[8];   // (10,500)
    const float* b3     = (const float*)d_in[9];   // (10,)
    float* out = (float*)d_out;                    // (256,10,100)

    // workspace layout (all sizes multiples of 256 B)
    const size_t O_W1S = 0;                         // 819200 B
    const size_t O_W2S = O_W1S + 819200;            // 1048576 B
    const size_t O_H1  = O_W2S + 1048576;           // 26214400 B
    const size_t O_SMP = O_H1  + 26214400;          // 51200000 B
    const size_t O_PSP = O_SMP + 51200000;          // 52428800 B
    const size_t O_CUR = O_PSP + 52428800;          // 1638400 B
    const size_t NEED  = O_CUR + 1638400;
    if (ws_size < NEED) return;

    char* ws = (char*)d_ws;
    _Float16* w1s    = (_Float16*)(ws + O_W1S);
    _Float16* w2s    = (_Float16*)(ws + O_W2S);
    _Float16* h1     = (_Float16*)(ws + O_H1);
    float*    sample = (float*)(ws + O_SMP);
    float*    psp    = (float*)(ws + O_PSP);
    float*    curb   = (float*)(ws + O_CUR);

    const double eta = 4.0;
    const float v0c = (float)(std::pow(eta, eta / (eta - 1.0)) / (eta - 1.0));
    const float am  = (float)std::exp(-1.0 / 4.0);

    swz_w1_kernel<<<100, 256, 0, stream>>>(w1, w1s);
    swz_w2_kernel<<<128, 256, 0, stream>>>(w2, w2s);
    gemm1_kernel<<<dim3(7, 256), 256, 0, stream>>>(x, b1, w1s, h1);
    gemm2_kernel<<<dim3(7, 256), 256, 0, stream>>>(h1, b2, w2s, eps, sample);
    axon_kernel<<<500, 256, 0, stream>>>(sample, alpha1, alpha2, v0c, psp);
    cur_kernel<<<3200, 256, 0, stream>>>(psp, w3, b3, curb);
    lif_kernel<<<10, 256, 0, stream>>>(curb, am, out);
}
